// StrengthNet_59304908423707
// MI455X (gfx1250) — matrix operand order repaired and checked by reading, not compiled, not run on hardware
//
#include <hip/hip_runtime.h>

// ---------------------------------------------------------------------------
// Set Transformer (ISAB x2 + PMA pool) forward for MI455X (gfx1250).
// All GEMMs run through v_wmma_f32_16x16x32_bf16 with fp32 accumulation.
// GEMM: 256x64 block tile, 8 waves, 32x64 per wave (2 WMMA row-tiles share
// each B fragment), double-buffered LDS staging software-pipelined so the
// global-load wait lands after the WMMA burst. fp32->bf16 conversion happens
// in the staging path. Softmax/LayerNorm are wave32 wave-per-row kernels.
// Workspace requirement: ~440 MiB (see offsets in kernel_launch).
// ---------------------------------------------------------------------------

typedef __bf16 bf16;
typedef bf16  v8bf  __attribute__((ext_vector_type(8)));
typedef bf16  v16bf __attribute__((ext_vector_type(16)));
typedef float v8f   __attribute__((ext_vector_type(8)));

#define BLK_M 256
#define BLK_N 64
#define BLK_K 32
#define LDS_P (BLK_K + 8)          // +16B pad per row: bank skew, keeps 16B align
#define GEMM_THREADS 256

#define SHUF16(lo, hi) __builtin_shufflevector(lo, hi, \
    0, 1, 2, 3, 4, 5, 6, 7, 8, 9, 10, 11, 12, 13, 14, 15)

// C[b][n][dout] = scale * sum_k A[b][n][k] * Bmat[k][dout] (+ bias[dout])
// transW=1: W given as [Dout][K] row-major (compute A@W^T)
// transW=0: W given as [K][Dout] row-major (compute A@W), transposed during
//           LDS staging. Requires Dout % 4 == 0. K must be a multiple of 32.
// Out-of-range tile rows/cols are loaded clamped (branch-free) and discarded
// by the guarded epilogue stores.
__global__ __launch_bounds__(GEMM_THREADS)
void gemm_bf16_wmma(const float* __restrict__ A, long long strideA,
                    const float* __restrict__ W, long long strideW,
                    const float* __restrict__ bias,
                    float* __restrict__ C, long long strideC,
                    int Nrows, int K, int Dout, float scale, int transW)
{
    __shared__ bf16 sA[2][BLK_M][LDS_P];   // 2 x 20 KB
    __shared__ bf16 sW[2][BLK_N][LDS_P];   // 2 x  5 KB   (always [out-col][k])

    const int tid = threadIdx.x;
    A += (size_t)blockIdx.z * strideA;
    W += (size_t)blockIdx.z * strideW;
    C += (size_t)blockIdx.z * strideC;

    const int rowBase = blockIdx.x * BLK_M;
    const int colBase = blockIdx.y * BLK_N;
    const int lane = tid & 31;
    const int wv   = tid >> 5;      // 8 waves, 32 rows each
    const int hf   = lane >> 4;
    const int l16  = lane & 15;

    v8f acc0[4], acc1[4];
#pragma unroll
    for (int n = 0; n < 4; ++n)
#pragma unroll
        for (int j = 0; j < 8; ++j) { acc0[n][j] = 0.f; acc1[n][j] = 0.f; }

    float4 ra[8];   // A tile: 256x32 = 2048 quads / 256 thr = 8 each
    float4 rw[2];   // W tile:  64x32 =  512 quads / 256 thr = 2 each

    // ---- staging: global -> regs (branch-free, clamped addresses) ----
    auto load_tiles = [&](int k0) {
#pragma unroll
        for (int i = 0; i < 8; ++i) {
            int q  = tid + i * GEMM_THREADS;
            int r  = q >> 3;
            int c4 = (q & 7) << 2;
            int gr = rowBase + r;
            gr = gr < Nrows ? gr : Nrows - 1;
            ra[i] = *(const float4*)(A + (size_t)gr * K + k0 + c4);
        }
        if (transW) {
#pragma unroll
            for (int i = 0; i < 2; ++i) {
                int q  = tid + i * GEMM_THREADS;
                int n  = q >> 3;
                int c4 = (q & 7) << 2;
                int gn = colBase + n;
                gn = gn < Dout ? gn : Dout - 1;
                rw[i] = *(const float4*)(W + (size_t)gn * K + k0 + c4);
            }
        } else {
#pragma unroll
            for (int i = 0; i < 2; ++i) {
                int q  = tid + i * GEMM_THREADS;
                int kk = q >> 4;            // 0..31
                int n4 = (q & 15) << 2;     // 0..60
                int gn = colBase + n4;
                gn = gn < Dout - 3 ? gn : Dout - 4;
                rw[i] = *(const float4*)(W + (size_t)(k0 + kk) * Dout + gn);
            }
        }
    };
    // ---- staging: regs -> LDS (bf16 convert) ----
    auto store_tiles = [&](int nb) {
#pragma unroll
        for (int i = 0; i < 8; ++i) {
            int q  = tid + i * GEMM_THREADS;
            int r  = q >> 3;
            int c4 = (q & 7) << 2;
            sA[nb][r][c4 + 0] = (bf16)ra[i].x; sA[nb][r][c4 + 1] = (bf16)ra[i].y;
            sA[nb][r][c4 + 2] = (bf16)ra[i].z; sA[nb][r][c4 + 3] = (bf16)ra[i].w;
        }
        if (transW) {
#pragma unroll
            for (int i = 0; i < 2; ++i) {
                int q  = tid + i * GEMM_THREADS;
                int n  = q >> 3;
                int c4 = (q & 7) << 2;
                sW[nb][n][c4 + 0] = (bf16)rw[i].x; sW[nb][n][c4 + 1] = (bf16)rw[i].y;
                sW[nb][n][c4 + 2] = (bf16)rw[i].z; sW[nb][n][c4 + 3] = (bf16)rw[i].w;
            }
        } else {
#pragma unroll
            for (int i = 0; i < 2; ++i) {
                int q  = tid + i * GEMM_THREADS;
                int kk = q >> 4;
                int n4 = (q & 15) << 2;
                sW[nb][n4 + 0][kk] = (bf16)rw[i].x; sW[nb][n4 + 1][kk] = (bf16)rw[i].y;
                sW[nb][n4 + 2][kk] = (bf16)rw[i].z; sW[nb][n4 + 3][kk] = (bf16)rw[i].w;
            }
        }
    };
    // ---- compute one 32-deep K-step from LDS buffer `buf` ----
    auto compute = [&](int buf) {
        const int r0 = (wv << 5) + l16;
        // A frag (ISA 16-bit A 16x32): lane = row, halves pick K chunks
        // {0..7,16..23} / {8..15,24..31}.
        v8bf alo0 = *(const v8bf*)&sA[buf][r0][hf << 3];
        v8bf ahi0 = *(const v8bf*)&sA[buf][r0][16 + (hf << 3)];
        v16bf a0 = SHUF16(alo0, ahi0);
        v8bf alo1 = *(const v8bf*)&sA[buf][r0 + 16][hf << 3];
        v8bf ahi1 = *(const v8bf*)&sA[buf][r0 + 16][16 + (hf << 3)];
        v16bf a1 = SHUF16(alo1, ahi1);
#pragma unroll
        for (int nt = 0; nt < 4; ++nt) {
            // B frag: lane = column, halves pick K {0..7,16..23} / {8..15,24..31}.
            const int bcol = (nt << 4) + l16;
            v8bf blo = *(const v8bf*)&sW[buf][bcol][hf << 3];
            v8bf bhi = *(const v8bf*)&sW[buf][bcol][16 + (hf << 3)];
            v16bf bb = SHUF16(blo, bhi);
            acc0[nt] = __builtin_amdgcn_wmma_f32_16x16x32_bf16(
                false, a0, false, bb, (short)0, acc0[nt], false, false);
            acc1[nt] = __builtin_amdgcn_wmma_f32_16x16x32_bf16(
                false, a1, false, bb, (short)0, acc1[nt], false, false);
        }
    };

    const int kSteps = K >> 5;
    // prologue: stage k-step 0
    load_tiles(0);
    store_tiles(0);
    __syncthreads();

    int buf = 0;
    for (int ks = 0; ks < kSteps; ++ks) {
        const bool hasNext = (ks + 1) < kSteps;
        if (hasNext) load_tiles((ks + 1) << 5);  // issue global loads early
        compute(buf);                            // WMMA burst hides load latency
        if (hasNext) store_tiles(buf ^ 1);       // waits loads, converts, ds_store
        __syncthreads();
        buf ^= 1;
    }

    // ---- epilogue: C 16x16 f32 layout: VGPR r -> M = r + 8*hf, N = l16 ----
#pragma unroll
    for (int nt = 0; nt < 4; ++nt) {
        int gc = colBase + (nt << 4) + l16;
        if (gc >= Dout) continue;
        float bv = bias ? bias[gc] : 0.f;
#pragma unroll
        for (int r = 0; r < 8; ++r) {
            int gr0 = rowBase + (wv << 5) + (hf << 3) + r;
            if (gr0 < Nrows)
                C[(size_t)gr0 * Dout + gc] = acc0[nt][r] * scale + bv;
            int gr1 = gr0 + 16;
            if (gr1 < Nrows)
                C[(size_t)gr1 * Dout + gc] = acc1[nt][r] * scale + bv;
        }
    }
}

// In-place row softmax, one wave32 per row.
__global__ __launch_bounds__(256)
void softmax_rows(float* __restrict__ S, int rows, int cols)
{
    int wv = threadIdx.x >> 5, lane = threadIdx.x & 31;
    int row = blockIdx.x * (blockDim.x >> 5) + wv;
    if (row >= rows) return;
    float* p = S + (size_t)row * cols;
    float mx = -3.0e38f;
    for (int c = lane; c < cols; c += 32) mx = fmaxf(mx, p[c]);
    for (int o = 16; o; o >>= 1) mx = fmaxf(mx, __shfl_xor(mx, o, 32));
    float sum = 0.f;
    for (int c = lane; c < cols; c += 32) sum += __expf(p[c] - mx);
    for (int o = 16; o; o >>= 1) sum += __shfl_xor(sum, o, 32);
    float inv = 1.f / sum;
    for (int c = lane; c < cols; c += 32) p[c] = __expf(p[c] - mx) * inv;
}

// Out[row] = LN( X[row % xMod] + (mode ? relu(Y[row]) : Y[row]) ) * g + b
// D fixed at 256; one wave32 per row (8 elements/lane).
__global__ __launch_bounds__(256)
void ln_rows256(const float* __restrict__ X, int xMod,
                const float* __restrict__ Y, float* __restrict__ O,
                const float* __restrict__ g, const float* __restrict__ bta,
                int rows, int mode)
{
    int wv = threadIdx.x >> 5, lane = threadIdx.x & 31;
    int row = blockIdx.x * (blockDim.x >> 5) + wv;
    if (row >= rows) return;
    const float* xr = X + (size_t)(row % xMod) * 256;
    const float* yr = Y + (size_t)row * 256;
    float v[8];
    float s = 0.f;
#pragma unroll
    for (int j = 0; j < 8; ++j) {
        int c = lane + 32 * j;
        float yv = yr[c];
        v[j] = xr[c] + (mode ? fmaxf(yv, 0.f) : yv);
        s += v[j];
    }
    for (int o = 16; o; o >>= 1) s += __shfl_xor(s, o, 32);
    float mean = s * (1.f / 256.f);
    float var = 0.f;
#pragma unroll
    for (int j = 0; j < 8; ++j) { float d = v[j] - mean; var += d * d; }
    for (int o = 16; o; o >>= 1) var += __shfl_xor(var, o, 32);
    float r = rsqrtf(var * (1.f / 256.f) + 1e-5f);
    float* orow = O + (size_t)row * 256;
#pragma unroll
    for (int j = 0; j < 8; ++j) {
        int c = lane + 32 * j;
        orow[c] = (v[j] - mean) * r * g[c] + bta[c];
    }
}

// out[row] = dot(P[row, 0:256], Wo[0:256]); one wave32 per row.
__global__ __launch_bounds__(256)
void dot_out_k(const float* __restrict__ P, const float* __restrict__ Wo,
               float* __restrict__ out, int rows)
{
    int wv = threadIdx.x >> 5, lane = threadIdx.x & 31;
    int row = blockIdx.x * (blockDim.x >> 5) + wv;
    if (row >= rows) return;
    float s = 0.f;
#pragma unroll
    for (int j = 0; j < 8; ++j) { int c = lane + 32 * j; s += P[(size_t)row * 256 + c] * Wo[c]; }
    for (int o = 16; o; o >>= 1) s += __shfl_xor(s, o, 32);
    if (lane == 0) out[row] = s;
}

// ---------------------------------------------------------------------------
// Host side
// ---------------------------------------------------------------------------

namespace {
enum { Fdim = 64, Ddim = 256, QDdim = 64, Mind = 32, Bsz = 64, Lseq = 2048,
       Ntok = Bsz * Lseq };

struct ABp { const float *WQ, *WK, *WV, *fcW, *fcb, *g0, *b0, *g1, *b1; };

ABp getAB(void* const* d_in, int base) {
    ABp p;
    p.WQ  = (const float*)d_in[base + 0];
    p.WK  = (const float*)d_in[base + 1];
    p.WV  = (const float*)d_in[base + 2];
    p.fcW = (const float*)d_in[base + 3];
    p.fcb = (const float*)d_in[base + 4];
    p.g0  = (const float*)d_in[base + 5];
    p.b0  = (const float*)d_in[base + 6];
    p.g1  = (const float*)d_in[base + 7];
    p.b1  = (const float*)d_in[base + 8];
    return p;
}

inline void gemm(hipStream_t st, const float* A, long long sA,
                 const float* W, long long sW, const float* bias,
                 float* C, long long sC, int Nrows, int K, int Dout,
                 float scale, int transW, int batch) {
    dim3 grid((Nrows + BLK_M - 1) / BLK_M, (Dout + BLK_N - 1) / BLK_N, batch);
    gemm_bf16_wmma<<<grid, GEMM_THREADS, 0, st>>>(A, sA, W, sW, bias, C, sC,
                                                  Nrows, K, Dout, scale, transW);
}

inline void softmax(hipStream_t st, float* S, int rows, int cols) {
    softmax_rows<<<(rows + 7) / 8, 256, 0, st>>>(S, rows, cols);
}

inline void ln(hipStream_t st, const float* X, int xMod, const float* Y,
               float* O, const float* g, const float* b, int rows, int mode) {
    ln_rows256<<<(rows + 7) / 8, 256, 0, st>>>(X, xMod, Y, O, g, b, rows, mode);
}
} // namespace

extern "C" void kernel_launch(void* const* d_in, const int* in_sizes, int n_in,
                              void* d_out, int out_size, void* d_ws, size_t ws_size,
                              hipStream_t stream)
{
    (void)in_sizes; (void)n_in; (void)out_size; (void)ws_size;

    // Input order = setup_inputs() insertion order (flattened):
    // 0:x 1:Win 2:isab0.i 3..11:isab0.ab0 12..20:isab0.ab1
    // 21:isab1.i 22..30:isab1.ab0 31..39:isab1.ab1
    // 40:pool.s 41..49:pool.ab 50:Wout 51:seqlen
    const float* x   = (const float*)d_in[0];
    const float* Win = (const float*)d_in[1];
    const float* iv[2] = { (const float*)d_in[2], (const float*)d_in[21] };
    ABp ab0s[2] = { getAB(d_in, 3),  getAB(d_in, 22) };
    ABp ab1s[2] = { getAB(d_in, 12), getAB(d_in, 31) };
    const float* ps   = (const float*)d_in[40];
    ABp pab           = getAB(d_in, 41);
    const float* Wout = (const float*)d_in[50];
    float* out = (float*)d_out;

    // Workspace layout (bytes). Needs ~440 MiB total.
    char* w = (char*)d_ws;
    float* H    = (float*)(w);                   // [N, 256]  134.2 MB
    float* T0   = (float*)(w + 134217728ULL);    // [N, 256]
    float* T1   = (float*)(w + 268435456ULL);    // [N, 256]
    float* KQ   = (float*)(w + 402653184ULL);    // [N, 64]    33.6 MB
    float* SC   = (float*)(w + 436207616ULL);    // scores     16.8 MB
    float* IND  = (float*)(w + 452984832ULL);    // [B*32,256]  2.1 MB
    float* TA   = (float*)(w + 455081984ULL);    // [B*32,256]
    float* TB   = (float*)(w + 457179136ULL);    // [B*32,256]
    float* QQ   = (float*)(w + 459276288ULL);    // [32,64]
    float* INDK = (float*)(w + 459341824ULL);    // [B*32,64]
    float* PO   = (float*)(w + 459931648ULL);    // [B,256]
    float* PH0  = (float*)(w + 460062720ULL);    // [B,256]
    float* PHR  = (float*)(w + 460193792ULL);    // [B,256]
    float* PLD  = (float*)(w + 460324864ULL);    // [B,256]

    const float invZ = 0.125f; // 1/sqrt(QD)

    // h = x @ Win^T : [N,64] x [256,64]^T -> [N,256]
    gemm(stream, x, 0, Win, 0, nullptr, H, 0, Ntok, Fdim, Ddim, 1.f, 1, 1);

    for (int l = 0; l < 2; ++l) {
        const ABp& p0 = ab0s[l];
        const ABp& p1 = ab1s[l];

        // ---- shared-query attention block: ind = AB(i, h) ----
        gemm(stream, H, 0, p0.WK, 0, nullptr, KQ, 0, Ntok, Ddim, QDdim, 1.f, 1, 1);
        gemm(stream, H, 0, p0.WV, 0, nullptr, T0, 0, Ntok, Ddim, Ddim, 1.f, 1, 1);
        gemm(stream, iv[l], 0, p0.WQ, 0, nullptr, QQ, 0, Mind, Ddim, QDdim, invZ, 1, 1);
        // scores[b] = qq @ k_b^T   (A shared across batch via strideA=0)
        gemm(stream, QQ, 0, KQ, (long long)Lseq * QDdim, nullptr,
             SC, (long long)Mind * Lseq, Mind, QDdim, Lseq, 1.f, 1, Bsz);
        softmax(stream, SC, Bsz * Mind, Lseq);
        // o[b] = a_b @ v_b
        gemm(stream, SC, (long long)Mind * Lseq, T0, (long long)Lseq * Ddim, nullptr,
             TA, (long long)Mind * Ddim, Mind, Lseq, Ddim, 1.f, 0, Bsz);
        ln(stream, iv[l], Mind, TA, TA, p0.g0, p0.b0, Bsz * Mind, 0);          // h0
        gemm(stream, TA, 0, p0.fcW, 0, p0.fcb, TB, 0, Bsz * Mind, Ddim, Ddim, 1.f, 1, 1);
        ln(stream, TA, Bsz * Mind, TB, IND, p0.g1, p0.b1, Bsz * Mind, 1);      // ind

        // ---- batched-query attention block: h = AB(h, ind) ----
        gemm(stream, H, 0, p1.WQ, 0, nullptr, KQ, 0, Ntok, Ddim, QDdim, invZ, 1, 1);
        gemm(stream, IND, 0, p1.WK, 0, nullptr, INDK, 0, Bsz * Mind, Ddim, QDdim, 1.f, 1, 1);
        gemm(stream, IND, 0, p1.WV, 0, nullptr, TB, 0, Bsz * Mind, Ddim, Ddim, 1.f, 1, 1);
        // scores[b] = qq_b @ k_b^T  ([L,64] x [32,64]^T -> [L,32])
        gemm(stream, KQ, (long long)Lseq * QDdim, INDK, (long long)Mind * QDdim, nullptr,
             SC, (long long)Lseq * Mind, Lseq, QDdim, Mind, 1.f, 1, Bsz);
        softmax(stream, SC, Ntok, Mind);
        // o[b] = a_b @ v_b  ([L,32] x [32,256])
        gemm(stream, SC, (long long)Lseq * Mind, TB, (long long)Mind * Ddim, nullptr,
             T1, (long long)Lseq * Ddim, Lseq, Mind, Ddim, 1.f, 0, Bsz);
        ln(stream, H, Ntok, T1, T1, p1.g0, p1.b0, Ntok, 0);                    // h0
        gemm(stream, T1, 0, p1.fcW, 0, p1.fcb, T0, 0, Ntok, Ddim, Ddim, 1.f, 1, 1);
        ln(stream, T1, Ntok, T0, H, p1.g1, p1.b1, Ntok, 1);                    // h
    }

    // ---- pool: 1-query shared attention + final projection ----
    gemm(stream, H, 0, pab.WK, 0, nullptr, KQ, 0, Ntok, Ddim, QDdim, 1.f, 1, 1);
    gemm(stream, H, 0, pab.WV, 0, nullptr, T0, 0, Ntok, Ddim, Ddim, 1.f, 1, 1);
    gemm(stream, ps, 0, pab.WQ, 0, nullptr, QQ, 0, 1, Ddim, QDdim, invZ, 1, 1);
    gemm(stream, QQ, 0, KQ, (long long)Lseq * QDdim, nullptr,
         SC, (long long)Lseq, 1, QDdim, Lseq, 1.f, 1, Bsz);
    softmax(stream, SC, Bsz, Lseq);
    gemm(stream, SC, (long long)Lseq, T0, (long long)Lseq * Ddim, nullptr,
         PO, (long long)Ddim, 1, Lseq, Ddim, 1.f, 0, Bsz);
    ln(stream, ps, 1, PO, PH0, pab.g0, pab.b0, Bsz, 0);
    gemm(stream, PH0, 0, pab.fcW, 0, pab.fcb, PHR, 0, Bsz, Ddim, Ddim, 1.f, 1, 1);
    ln(stream, PH0, Bsz, PHR, PLD, pab.g1, pab.b1, Bsz, 1);
    dot_out_k<<<(Bsz + 7) / 8, 256, 0, stream>>>(PLD, Wout, out, Bsz);
}
